// OmniAID_24618752540910
// MI455X (gfx1250) — compile-verified, matrix operand order checked by reading, not run on hardware
//
#include <hip/hip_runtime.h>
#include <hip/hip_bf16.h>
#include <cstdint>

#define B_  32
#define S_  257
#define D_  1024
#define H_  16
#define E_  8
#define R_  64
#define K_  2
#define FF_ 4096
#define HR_ 256
#define HD_ 64
#define BS_ (B_ * S_)

typedef __bf16 bf16;
typedef __attribute__((ext_vector_type(16))) __bf16 v16bf;
typedef __attribute__((ext_vector_type(8)))  float  v8f;

#define LDSA 40   // LDS row stride (bf16 elems): 80B, 16B-aligned rows
#define LDSB 40

// ---------------------------------------------------------------------------
// CDNA5 primitives
// ---------------------------------------------------------------------------
__device__ inline v8f wmma_bf16(v16bf a, v16bf b, v8f c) {
  return __builtin_amdgcn_wmma_f32_16x16x32_bf16(
      /*neg_a=*/false, a, /*neg_b=*/false, b,
      /*c_mod=*/(short)0, c, /*reuse_a=*/false, /*reuse_b=*/false);
}

// Async 16B global -> LDS copy (ASYNCcnt-tracked tensor/async path).
__device__ inline void async_b128(uint32_t lds_byte_addr, const void* gaddr) {
  asm volatile("global_load_async_to_lds_b128 %0, %1, off"
               :: "v"(lds_byte_addr), "v"(gaddr) : "memory");
}

template <int N>
__device__ inline void wait_async() {
  asm volatile("s_wait_asynccnt %0" :: "i"(N) : "memory");
}

// Low 32 bits of a generic pointer to LDS = LDS byte address (aperture is in
// the upper 32 bits on gfx1250).
__device__ inline uint32_t lds_addr(const void* p) {
  return (uint32_t)(uintptr_t)p;
}

// Load a 16x32 bf16 fragment (A or B) from LDS in the CDNA5 wave32 layout:
// lanes 0-15 -> rows 0-15, k in {0..7, 16..23}; lanes 16-31 -> k base +8.
__device__ inline v16bf frag_ld(const bf16* lds, int stride, int row, int koff) {
  const int lane = threadIdx.x & 31;
  const bf16* p = lds + (row + (lane & 15)) * stride + koff + ((lane < 16) ? 0 : 8);
  v16bf f;
#pragma unroll
  for (int i = 0; i < 8; ++i) { f[i] = p[i]; f[i + 8] = p[i + 16]; }
  return f;
}

__device__ inline void acc_zero(v8f acc[2][2]) {
#pragma unroll
  for (int a = 0; a < 2; ++a)
#pragma unroll
    for (int b = 0; b < 2; ++b)
#pragma unroll
      for (int j = 0; j < 8; ++j) acc[a][b][j] = 0.f;
}

// ---------------------------------------------------------------------------
// Stage one 64x32 A tile + 128x32 B tile (bf16) into LDS via async copies.
// OOB rows are clamped to a valid row: garbage only reaches epilogue-masked
// outputs (col n of D depends only on row n of B; row m only on row m of A).
// Exactly 3 async instructions per wave per stage (1 for A, 2 for B).
// ---------------------------------------------------------------------------
__device__ inline void stage_tile(const bf16* __restrict__ A, long lda, long m0, long Mmax,
                                  const bf16* __restrict__ Bw, long ldb, long n0, long Nmax,
                                  int k0, bf16* sA, bf16* sB) {
  const int t = threadIdx.x;
  {
    const int r = t >> 2, ch = t & 3;             // 64 rows x 4 chunks of 8 bf16
    long gm = m0 + r; if (gm >= Mmax) gm = Mmax - 1;
    async_b128(lds_addr(sA + r * LDSA + ch * 8), A + gm * lda + k0 + ch * 8);
  }
#pragma unroll
  for (int q = 0; q < 2; ++q) {                    // 128 rows x 4 chunks
    const int idx = t + q * 256;
    const int n = idx >> 2, ch = idx & 3;
    long gn = n0 + n; if (gn >= Nmax) gn = Nmax - 1;
    async_b128(lds_addr(sB + n * LDSB + ch * 8), Bw + gn * ldb + k0 + ch * 8);
  }
}

// ---------------------------------------------------------------------------
// 64x128 block-tile WMMA GEMM accumulator, double-buffered async pipeline.
// A: bf16 [M, Kd] row-major.  B: bf16 [N, Kd] row-major ([out, in]).
// 256 threads = 8 waves (2m x 4n); each wave owns 32x32 (2x2 v8f accs)
// -> 4 v_wmma_f32_16x16x32_bf16 per wave per 32-K step.
// ---------------------------------------------------------------------------
__device__ void gemm_accum(v8f acc[2][2],
                           const bf16* __restrict__ A, long lda, long m0, long Mmax,
                           const bf16* __restrict__ Bw, long ldb, long n0, long Nmax,
                           int Kd, bf16* sAbuf, bf16* sBbuf) {
  const int wave = threadIdx.x >> 5;
  const int wm = wave & 1, wn = wave >> 1;
  const int nk = Kd >> 5;

  stage_tile(A, lda, m0, Mmax, Bw, ldb, n0, Nmax, 0, sAbuf, sBbuf);

  for (int kt = 0; kt < nk; ++kt) {
    bf16* cA = sAbuf + (kt & 1) * (64 * LDSA);
    bf16* cB = sBbuf + (kt & 1) * (128 * LDSB);
    if (kt + 1 < nk) {
      bf16* nA = sAbuf + ((kt + 1) & 1) * (64 * LDSA);
      bf16* nB = sBbuf + ((kt + 1) & 1) * (128 * LDSB);
      stage_tile(A, lda, m0, Mmax, Bw, ldb, n0, Nmax, (kt + 1) * 32, nA, nB);
      wait_async<3>();   // current stage complete; next (3 ops) still in flight
    } else {
      wait_async<0>();
    }
    __syncthreads();

    v16bf af0 = frag_ld(cA, LDSA, wm * 32,      0);
    v16bf af1 = frag_ld(cA, LDSA, wm * 32 + 16, 0);
    v16bf bf0 = frag_ld(cB, LDSB, wn * 32,      0);
    v16bf bf1 = frag_ld(cB, LDSB, wn * 32 + 16, 0);
    acc[0][0] = wmma_bf16(af0, bf0, acc[0][0]);
    acc[0][1] = wmma_bf16(af0, bf1, acc[0][1]);
    acc[1][0] = wmma_bf16(af1, bf0, acc[1][0]);
    acc[1][1] = wmma_bf16(af1, bf1, acc[1][1]);
    __syncthreads();
  }
}

// Epilogue iterator: yields (m, n, v) per lane-owned accumulator element.
// C/D layout: VGPR j -> M = j (lanes 0-15) or j+8 (lanes 16-31); N = lane&15.
#define EPI_BEGIN                                                           \
  {                                                                         \
    const int lane_ = threadIdx.x & 31;                                     \
    const int wave_ = threadIdx.x >> 5;                                     \
    const int wm_ = wave_ & 1, wn_ = wave_ >> 1;                            \
    _Pragma("unroll") for (int sm = 0; sm < 2; ++sm)                        \
    _Pragma("unroll") for (int sn = 0; sn < 2; ++sn)                        \
    _Pragma("unroll") for (int j = 0; j < 8; ++j) {                         \
      const long m = m0 + wm_ * 32 + sm * 16 + j + ((lane_ >= 16) ? 8 : 0); \
      const long n = n0 + wn_ * 32 + sn * 16 + (lane_ & 15);                \
      const float v = acc[sm][sn][j];
#define EPI_END } }

#define GEMM_SHARED                                   \
  __shared__ __align__(16) bf16 sA[2 * 64 * LDSA];    \
  __shared__ __align__(16) bf16 sB[2 * 128 * LDSB];

// ---------------------------------------------------------------------------
// Weight pre-conversion: f32 -> bf16 (copy) and f32 -> bf16 (transpose).
// ---------------------------------------------------------------------------
__global__ void k_cvt(const float* __restrict__ s, bf16* __restrict__ d, long n) {
  const long i = ((long)blockIdx.x * 256 + threadIdx.x) * 8;
  if (i + 8 <= n) {
#pragma unroll
    for (int j = 0; j < 8; ++j) d[i + j] = (bf16)s[i + j];
  } else {
    for (long j = i; j < n; ++j) d[j] = (bf16)s[j];
  }
}

// src: [RW, CL] f32  ->  dst: [CL, RW] bf16   (RW, CL multiples of 32)
__global__ void k_cvt_t(const float* __restrict__ src, bf16* __restrict__ dst,
                        int RW, int CL) {
  __shared__ float t[32][33];
  const int r0 = blockIdx.y * 32, c0 = blockIdx.x * 32;
  const int tr = threadIdx.x & 31, tc = threadIdx.x >> 5;  // 32 x 8
#pragma unroll
  for (int i = 0; i < 32; i += 8)
    t[tc + i][tr] = src[(long)(r0 + tc + i) * CL + c0 + tr];
  __syncthreads();
#pragma unroll
  for (int i = 0; i < 32; i += 8)
    dst[(long)(c0 + tc + i) * RW + r0 + tr] = (bf16)t[tr][tc + i];
}

// ---------------------------------------------------------------------------
// Gating: hcls = relu(h[b,0]@Wg1+bg1); logits = hcls@Wg2+bg2; top-2 softmax.
// ---------------------------------------------------------------------------
__global__ void k_gate(const float* __restrict__ hs, const float* __restrict__ Wg1,
                       const float* __restrict__ bg1, const float* __restrict__ Wg2,
                       const float* __restrict__ bg2, int* __restrict__ topi,
                       float* __restrict__ gates) {
  const int b = blockIdx.x;
  const int t = threadIdx.x;  // 256
  __shared__ float h0[D_];
  __shared__ float hc[HR_];
  __shared__ float lg[E_];
  for (int i = t; i < D_; i += 256) h0[i] = hs[(long)b * S_ * D_ + i];
  __syncthreads();
  float a = bg1[t];
  for (int d = 0; d < D_; ++d) a += h0[d] * Wg1[(long)d * HR_ + t];
  hc[t] = a > 0.f ? a : 0.f;
  __syncthreads();
  if (t < E_) {
    float l = bg2[t];
    for (int r = 0; r < HR_; ++r) l += hc[r] * Wg2[r * E_ + t];
    lg[t] = l;
  }
  __syncthreads();
  if (t == 0) {
    int i0 = 0;
    for (int e = 1; e < E_; ++e) if (lg[e] > lg[i0]) i0 = e;
    int i1 = (i0 == 0) ? 1 : 0;
    for (int e = 0; e < E_; ++e) if (e != i0 && lg[e] > lg[i1]) i1 = e;
    const float mx = lg[i0];
    const float e0 = __expf(lg[i0] - mx), e1 = __expf(lg[i1] - mx);
    const float inv = 1.f / (e0 + e1);
    topi[b * 2] = i0;  topi[b * 2 + 1] = i1;
    gates[b * 2] = e0 * inv;  gates[b * 2 + 1] = e1 * inv;
  }
}

// ---------------------------------------------------------------------------
// LayerNorm over last dim (1024), emits bf16.
// ---------------------------------------------------------------------------
__global__ void k_ln(const float* __restrict__ X, const float* __restrict__ g,
                     const float* __restrict__ bta, bf16* __restrict__ out) {
  const long row = blockIdx.x;
  const int t = threadIdx.x;  // 256
  const float* x = X + row * D_;
  __shared__ float red[256];
  float v4[4];
  float s = 0.f;
#pragma unroll
  for (int i = 0; i < 4; ++i) { v4[i] = x[t + i * 256]; s += v4[i]; }
  red[t] = s; __syncthreads();
  for (int o = 128; o > 0; o >>= 1) { if (t < o) red[t] += red[t + o]; __syncthreads(); }
  const float mean = red[0] * (1.f / D_);
  __syncthreads();
  s = 0.f;
#pragma unroll
  for (int i = 0; i < 4; ++i) { const float d = v4[i] - mean; s += d * d; }
  red[t] = s; __syncthreads();
  for (int o = 128; o > 0; o >>= 1) { if (t < o) red[t] += red[t + o]; __syncthreads(); }
  const float inv = rsqrtf(red[0] * (1.f / D_) + 1e-5f);
#pragma unroll
  for (int i = 0; i < 4; ++i) {
    const int c = t + i * 256;
    out[row * D_ + c] = (bf16)((v4[i] - mean) * inv * g[c] + bta[c]);
  }
}

// ---------------------------------------------------------------------------
// Main q/k/v projections: QKV[p] = (xbf @ Wm[p]^T + b[p]) * alpha_p  (bf16)
// ---------------------------------------------------------------------------
__global__ void k_proj_main(const bf16* __restrict__ X, const bf16* __restrict__ Wm,
                            const float* __restrict__ bias, float scale,
                            bf16* __restrict__ QKV) {
  GEMM_SHARED
  const int p = blockIdx.z;
  const long m0 = (long)blockIdx.y * 64;
  const long n0 = (long)blockIdx.x * 128;
  v8f acc[2][2]; acc_zero(acc);
  gemm_accum(acc, X, D_, m0, BS_, Wm + (long)p * D_ * D_, D_, n0, D_, D_, sA, sB);
  const float alpha = (p == 0) ? scale : 1.f;
  const float* bp = bias + p * D_;
  bf16* out = QKV + (long)p * BS_ * D_;
  EPI_BEGIN
    if (m < BS_) out[m * D_ + n] = (bf16)((v + bp[n]) * alpha);
  EPI_END
}

// ---------------------------------------------------------------------------
// xv = (x_b @ V[e]^T) * S[e] * gate   -> bf16 [B*K, S, R]
// ---------------------------------------------------------------------------
__global__ void k_xv(const bf16* __restrict__ X, const bf16* __restrict__ Vbase,
                     const float* __restrict__ Sbase, const int* __restrict__ topi,
                     const float* __restrict__ gates, bf16* __restrict__ xv) {
  GEMM_SHARED
  const int z = blockIdx.z;
  const int b = z >> 1, ks = z & 1;
  const int e = topi[b * 2 + ks];
  const float gate = gates[b * 2 + ks];
  const long m0 = (long)blockIdx.y * 64;
  const long n0 = 0;
  v8f acc[2][2]; acc_zero(acc);
  gemm_accum(acc, X + (long)b * S_ * D_, D_, m0, S_,
             Vbase + (long)e * R_ * D_, D_, n0, R_, D_, sA, sB);
  bf16* out = xv + (long)(b * 2 + ks) * S_ * R_;
  const float* sp = Sbase + e * R_;
  EPI_BEGIN
    if (m < S_ && n < R_) out[m * R_ + n] = (bf16)(v * sp[(int)n] * gate);
  EPI_END
}

// ---------------------------------------------------------------------------
// eo = sum_k xv_k @ U[e_k]^T, accumulated in WMMA accs; added into output
// (bf16 r-m-w for q/k/v, or f32 += for hmid).
// ---------------------------------------------------------------------------
__global__ void k_eo(const bf16* __restrict__ xv, const bf16* __restrict__ Ubase,
                     const int* __restrict__ topi, float alpha,
                     bf16* __restrict__ out_bf, float* __restrict__ out_f32) {
  GEMM_SHARED
  const int b = blockIdx.z;
  const long m0 = (long)blockIdx.y * 64;
  const long n0 = (long)blockIdx.x * 128;
  v8f acc[2][2]; acc_zero(acc);
  for (int ks = 0; ks < 2; ++ks) {
    const int e = topi[b * 2 + ks];
    gemm_accum(acc, xv + (long)(b * 2 + ks) * S_ * R_, R_, m0, S_,
               Ubase + (long)e * D_ * R_, R_, n0, D_, R_, sA, sB);
  }
  EPI_BEGIN
    if (m < S_) {
      const long idx = ((long)b * S_ + m) * D_ + n;
      if (out_f32) {
        out_f32[idx] += v * alpha;
      } else {
        const float cur = (float)out_bf[idx];
        out_bf[idx] = (bf16)(cur + v * alpha);
      }
    }
  EPI_END
}

// ---------------------------------------------------------------------------
// Attention: per (b, h, 32-row q tile). Scores via WMMA (Kd=64), row softmax
// over S padded to 288, O = P @ V via WMMA (9 K-steps of 32). Scale already
// folded into q.
// ---------------------------------------------------------------------------
__global__ void k_attn(const bf16* __restrict__ QKV, bf16* __restrict__ obf) {
  const bf16* Q  = QKV;
  const bf16* Kk = QKV + (long)BS_ * D_;
  const bf16* Vv = QKV + 2L * BS_ * D_;
  const int b = blockIdx.z, h = blockIdx.y;
  const long m0 = (long)blockIdx.x * 32;
  const int tid = threadIdx.x, lane = tid & 31, wave = tid >> 5;

  __shared__ __align__(16) bf16 qs[32 * 72];
  __shared__ float sc[32 * 296];
  __shared__ bf16  ps[32 * 296];

  // load 32x64 q strip
  {
    const int r = tid >> 3;         // 0..31
    const int c = (tid & 7) * 8;    // 0..56
    const long s = m0 + r;
#pragma unroll
    for (int i = 0; i < 8; ++i)
      qs[r * 72 + c + i] = (s < S_) ? Q[((long)b * S_ + s) * D_ + h * HD_ + c + i]
                                    : (bf16)0.f;
  }
  __syncthreads();

  // scores: 18 n-tiles of 16 cols (288 padded), distributed over 8 waves
  for (int nt = wave; nt < 18; nt += 8) {
    v8f acc2[2];
#pragma unroll
    for (int sm = 0; sm < 2; ++sm)
#pragma unroll
      for (int j = 0; j < 8; ++j) acc2[sm][j] = 0.f;
#pragma unroll
    for (int kst = 0; kst < 2; ++kst) {
      v16bf bfv;
      const int srow = nt * 16 + (lane & 15);
      const int kb = kst * 32 + ((lane < 16) ? 0 : 8);
      const bool ok = srow < S_;
      const bf16* kp = Kk + ((long)b * S_ + (ok ? srow : 0)) * D_ + h * HD_ + kb;
#pragma unroll
      for (int i = 0; i < 8; ++i) {
        bfv[i]     = ok ? kp[i]      : (bf16)0.f;
        bfv[i + 8] = ok ? kp[i + 16] : (bf16)0.f;
      }
      v16bf a0 = frag_ld(qs, 72, 0,  kst * 32);
      v16bf a1 = frag_ld(qs, 72, 16, kst * 32);
      acc2[0] = wmma_bf16(a0, bfv, acc2[0]);
      acc2[1] = wmma_bf16(a1, bfv, acc2[1]);
    }
#pragma unroll
    for (int sm = 0; sm < 2; ++sm)
#pragma unroll
      for (int j = 0; j < 8; ++j) {
        const int m = sm * 16 + j + ((lane >= 16) ? 8 : 0);
        const int n = nt * 16 + (lane & 15);
        sc[m * 296 + n] = (n < S_) ? acc2[sm][j] : -1e30f;
      }
  }
  __syncthreads();

  // softmax: each wave owns 4 rows; lanes stride the 288 columns
  for (int rr = 0; rr < 4; ++rr) {
    const int row = wave * 4 + rr;
    float mx = -1e30f;
    for (int c = lane; c < 288; c += 32) mx = fmaxf(mx, sc[row * 296 + c]);
#pragma unroll
    for (int o = 16; o > 0; o >>= 1) mx = fmaxf(mx, __shfl_xor(mx, o, 32));
    float sum = 0.f;
    for (int c = lane; c < 288; c += 32) sum += __expf(sc[row * 296 + c] - mx);
#pragma unroll
    for (int o = 16; o > 0; o >>= 1) sum += __shfl_xor(sum, o, 32);
    const float inv = (sum > 0.f) ? 1.f / sum : 0.f;
    for (int c = lane; c < 288; c += 32)
      ps[row * 296 + c] = (bf16)(__expf(sc[row * 296 + c] - mx) * inv);
  }
  __syncthreads();

  // O = P(32x288) @ V(288x64): each wave owns one 16x16 output tile
  {
    const int wm = wave & 1, wn = wave >> 1;
    v8f acc;
#pragma unroll
    for (int j = 0; j < 8; ++j) acc[j] = 0.f;
    for (int kst = 0; kst < 9; ++kst) {
      v16bf af = frag_ld(ps, 296, wm * 16, kst * 32);
      v16bf bfv;
      const int kb = kst * 32 + ((lane < 16) ? 0 : 8);
      const int ncol = wn * 16 + (lane & 15);
#pragma unroll
      for (int i = 0; i < 8; ++i) {
        const int s0 = kb + i, s1 = kb + 16 + i;
        bfv[i]     = (s0 < S_) ? Vv[((long)b * S_ + s0) * D_ + h * HD_ + ncol] : (bf16)0.f;
        bfv[i + 8] = (s1 < S_) ? Vv[((long)b * S_ + s1) * D_ + h * HD_ + ncol] : (bf16)0.f;
      }
      acc = wmma_bf16(af, bfv, acc);
    }
#pragma unroll
    for (int j = 0; j < 8; ++j) {
      const long m = m0 + wm * 16 + j + ((lane >= 16) ? 8 : 0);
      const int n = wn * 16 + (lane & 15);
      if (m < S_) obf[((long)b * S_ + m) * D_ + h * HD_ + n] = (bf16)acc[j];
    }
  }
}

// ---------------------------------------------------------------------------
// Output projection main path: hmid = hidden + obf @ Wm3^T + b3   (f32)
// ---------------------------------------------------------------------------
__global__ void k_outproj(const bf16* __restrict__ X, const bf16* __restrict__ Wm3,
                          const float* __restrict__ bias3, const float* __restrict__ hidden,
                          float* __restrict__ hmid) {
  GEMM_SHARED
  const long m0 = (long)blockIdx.y * 64;
  const long n0 = (long)blockIdx.x * 128;
  v8f acc[2][2]; acc_zero(acc);
  gemm_accum(acc, X, D_, m0, BS_, Wm3, D_, n0, D_, D_, sA, sB);
  EPI_BEGIN
    if (m < BS_) {
      const long idx = m * D_ + n;
      hmid[idx] = hidden[idx] + v + bias3[n];
    }
  EPI_END
}

// ---------------------------------------------------------------------------
// MLP fc1 with quick-GELU -> bf16 [BS, FF]; W1t is pre-transposed [FF, D]
// ---------------------------------------------------------------------------
__global__ void k_fc1(const bf16* __restrict__ X, const bf16* __restrict__ W1t,
                      const float* __restrict__ b1, bf16* __restrict__ fbf) {
  GEMM_SHARED
  const long m0 = (long)blockIdx.y * 64;
  const long n0 = (long)blockIdx.x * 128;
  v8f acc[2][2]; acc_zero(acc);
  gemm_accum(acc, X, D_, m0, BS_, W1t, D_, n0, FF_, D_, sA, sB);
  EPI_BEGIN
    if (m < BS_) {
      const float f = v + b1[n];
      const float gl = f / (1.f + __expf(-1.702f * f));
      fbf[m * FF_ + n] = (bf16)gl;
    }
  EPI_END
}

// ---------------------------------------------------------------------------
// MLP fc2 + residual -> f32 output; W2t is pre-transposed [D, FF]
// ---------------------------------------------------------------------------
__global__ void k_fc2(const bf16* __restrict__ F, const bf16* __restrict__ W2t,
                      const float* __restrict__ b2, const float* __restrict__ hmid,
                      float* __restrict__ out) {
  GEMM_SHARED
  const long m0 = (long)blockIdx.y * 64;
  const long n0 = (long)blockIdx.x * 128;
  v8f acc[2][2]; acc_zero(acc);
  gemm_accum(acc, F, FF_, m0, BS_, W2t, FF_, n0, D_, FF_, sA, sB);
  EPI_BEGIN
    if (m < BS_) {
      const long idx = m * D_ + n;
      out[idx] = hmid[idx] + v + b2[n];
    }
  EPI_END
}

// ---------------------------------------------------------------------------
// Host orchestration
// ---------------------------------------------------------------------------
extern "C" void kernel_launch(void* const* d_in, const int* in_sizes, int n_in,
                              void* d_out, int out_size, void* d_ws, size_t ws_size,
                              hipStream_t stream) {
  (void)in_sizes; (void)n_in; (void)out_size; (void)ws_size;
  const float* hs     = (const float*)d_in[0];
  const float* W_main = (const float*)d_in[1];
  const float* U_all  = (const float*)d_in[2];
  const float* S_all  = (const float*)d_in[3];
  const float* V_all  = (const float*)d_in[4];
  const float* b_proj = (const float*)d_in[5];
  const float* Wg1    = (const float*)d_in[6];
  const float* bg1    = (const float*)d_in[7];
  const float* Wg2    = (const float*)d_in[8];
  const float* bg2    = (const float*)d_in[9];
  const float* ln1g   = (const float*)d_in[10];
  const float* ln1b   = (const float*)d_in[11];
  const float* ln2g   = (const float*)d_in[12];
  const float* ln2b   = (const float*)d_in[13];
  const float* W_fc1  = (const float*)d_in[14];
  const float* b_fc1  = (const float*)d_in[15];
  const float* W_fc2  = (const float*)d_in[16];
  const float* b_fc2  = (const float*)d_in[17];
  float* out = (float*)d_out;

  char* w = (char*)d_ws;
  auto alloc = [&](size_t bytes) {
    char* p = w;
    w += (bytes + 255) & ~(size_t)255;
    return p;
  };
  int*   topi  = (int*)  alloc((size_t)B_ * K_ * sizeof(int));
  float* gates = (float*)alloc((size_t)B_ * K_ * sizeof(float));
  bf16*  xbf   = (bf16*) alloc((size_t)BS_ * D_ * 2);
  bf16*  qkv   = (bf16*) alloc((size_t)3 * BS_ * D_ * 2);
  bf16*  xv    = (bf16*) alloc((size_t)B_ * K_ * S_ * R_ * 2);
  bf16*  obf   = (bf16*) alloc((size_t)BS_ * D_ * 2);
  float* hmid  = (float*)alloc((size_t)BS_ * D_ * 4);
  bf16*  x2bf  = (bf16*) alloc((size_t)BS_ * D_ * 2);
  bf16*  fbf   = (bf16*) alloc((size_t)BS_ * FF_ * 2);
  // pre-converted bf16 weights
  bf16*  Wm_bf = (bf16*) alloc((size_t)4 * D_ * D_ * 2);
  bf16*  U_bf  = (bf16*) alloc((size_t)4 * E_ * D_ * R_ * 2);
  bf16*  V_bf  = (bf16*) alloc((size_t)4 * E_ * R_ * D_ * 2);
  bf16*  W1t   = (bf16*) alloc((size_t)FF_ * D_ * 2);
  bf16*  W2t   = (bf16*) alloc((size_t)D_ * FF_ * 2);

  const float scale = 0.125f;  // HD^-0.5 = 64^-0.5

  // weight conversion (one pass; all GEMMs then consume bf16, async-stageable)
  k_cvt<<<(4L * D_ * D_) / 2048, 256, 0, stream>>>(W_main, Wm_bf, 4L * D_ * D_);
  k_cvt<<<(4L * E_ * D_ * R_) / 2048, 256, 0, stream>>>(U_all, U_bf, 4L * E_ * D_ * R_);
  k_cvt<<<(4L * E_ * R_ * D_) / 2048, 256, 0, stream>>>(V_all, V_bf, 4L * E_ * R_ * D_);
  { dim3 g(FF_ / 32, D_ / 32); k_cvt_t<<<g, 256, 0, stream>>>(W_fc1, W1t, D_, FF_); }
  { dim3 g(D_ / 32, FF_ / 32); k_cvt_t<<<g, 256, 0, stream>>>(W_fc2, W2t, FF_, D_); }

  k_gate<<<B_, 256, 0, stream>>>(hs, Wg1, bg1, Wg2, bg2, topi, gates);
  k_ln<<<BS_, 256, 0, stream>>>(hs, ln1g, ln1b, xbf);

  {
    dim3 g(D_ / 128, (BS_ + 63) / 64, 3);
    k_proj_main<<<g, 256, 0, stream>>>(xbf, Wm_bf, b_proj, scale, qkv);
  }
  for (int p = 0; p < 3; ++p) {
    dim3 gx(1, (S_ + 63) / 64, B_ * K_);
    k_xv<<<gx, 256, 0, stream>>>(xbf, V_bf + (size_t)p * E_ * R_ * D_,
                                 S_all + p * E_ * R_, topi, gates, xv);
    dim3 ge(D_ / 128, (S_ + 63) / 64, B_);
    k_eo<<<ge, 256, 0, stream>>>(xv, U_bf + (size_t)p * E_ * D_ * R_, topi,
                                 (p == 0) ? scale : 1.f,
                                 qkv + (size_t)p * BS_ * D_, nullptr);
  }
  {
    dim3 ga((S_ + 31) / 32, H_, B_);
    k_attn<<<ga, 256, 0, stream>>>(qkv, obf);
  }
  {
    dim3 g(D_ / 128, (BS_ + 63) / 64);
    k_outproj<<<g, 256, 0, stream>>>(obf, Wm_bf + 3L * D_ * D_, b_proj + 3 * D_,
                                     hs, hmid);
  }
  {
    dim3 gx(1, (S_ + 63) / 64, B_ * K_);
    k_xv<<<gx, 256, 0, stream>>>(obf, V_bf + 3L * E_ * R_ * D_,
                                 S_all + 3 * E_ * R_, topi, gates, xv);
    dim3 ge(D_ / 128, (S_ + 63) / 64, B_);
    k_eo<<<ge, 256, 0, stream>>>(xv, U_bf + 3L * E_ * D_ * R_, topi, 1.f,
                                 nullptr, hmid);
  }
  k_ln<<<BS_, 256, 0, stream>>>(hmid, ln2g, ln2b, x2bf);
  {
    dim3 g(FF_ / 128, (BS_ + 63) / 64);
    k_fc1<<<g, 256, 0, stream>>>(x2bf, W1t, b_fc1, fbf);
  }
  {
    dim3 g(D_ / 128, (BS_ + 63) / 64);
    k_fc2<<<g, 256, 0, stream>>>(fbf, W2t, b_fc2, hmid, out);
  }
}
